// FFM_37864431682361
// MI455X (gfx1250) — compile-verified, matrix operand order checked by reading, not run on hardware
//
#include <hip/hip_runtime.h>
#include <math.h>

typedef __attribute__((ext_vector_type(16))) _Float16 v16h;
typedef __attribute__((ext_vector_type(8)))  _Float16 v8h;
typedef __attribute__((ext_vector_type(4)))  _Float16 v4h;
typedef __attribute__((ext_vector_type(8)))  float    v8f;
typedef __attribute__((ext_vector_type(4)))  float    v4f;
typedef __attribute__((ext_vector_type(4)))  unsigned int u32x4;
typedef __attribute__((ext_vector_type(8)))  int      i32x8;
typedef __attribute__((ext_vector_type(4)))  int      i32x4;

union ABu { v16h v; v8h p[2]; };

#define DDIM 1024
#define BDIM 16384

// ---------------------------------------------------------------------------
// TDM: 1-D tensor load, `n8` 8-byte elements, global -> LDS (per-wave DMA).
// D# packing per CDNA5 ISA:
//   group0: [1:0]=count=1, [63:32]=lds_addr, [120:64]=global_addr, [127:126]=2
//   group1: [17:16]=data_size(3 => 8B), [79:48]=tensor_dim0, [111:80]=tensor_dim1,
//           [127:112]=tile_dim0, tile_dim1/2=0 (unused), [207:160]=dim0_stride
// This toolchain exposes the 6-arg builtin:
//   (u32x4 g0, i32x8 g1, i32x4 g2, i32x4 g3, i32x8 g4, i32 cpol)
// ---------------------------------------------------------------------------
__device__ __forceinline__ void tdm_load_1d(unsigned int ldsOff,
                                            const void* gptr,
                                            unsigned int n8) {
    unsigned long long ga = (unsigned long long)(size_t)gptr;
    u32x4 g0;
    g0[0] = 1u;                                            // count=1, user D#
    g0[1] = ldsOff;                                        // lds_addr (bytes)
    g0[2] = (unsigned int)ga;                              // global_addr[31:0]
    g0[3] = (unsigned int)((ga >> 32) & 0x1FFFFFFu) | (2u << 30); // [56:32]|type=2
    i32x8 g1;
    g1[0] = (int)(3u << 16);                               // data_size=8B, mask=0
    g1[1] = (int)((n8 & 0xFFFFu) << 16);                   // tensor_dim0[15:0]
    g1[2] = (int)(((n8 >> 16) & 0xFFFFu) | (1u << 16));    // dim0[31:16], dim1=1
    g1[3] = (int)((n8 & 0xFFFFu) << 16);                   // tile_dim0
    g1[4] = 0;                                             // tile_dim1/2 unused
    g1[5] = (int)n8;                                       // dim0_stride (benign)
    g1[6] = 0;
    g1[7] = 0;
    i32x4 z4 = {0, 0, 0, 0};                               // <=2D: groups 2/3 null
    i32x8 z8 = {0, 0, 0, 0, 0, 0, 0, 0};
    __builtin_amdgcn_tensor_load_to_lds(g0, g1, z4, z4, z8, 0);
}

// ---------------------------------------------------------------------------
// Zero the temp accumulator (graph-capture safe, deterministic per launch).
// ---------------------------------------------------------------------------
__global__ __launch_bounds__(256) void ffm_zero(float* __restrict__ t) {
    t[blockIdx.x * 256 + threadIdx.x] = 0.0f;
}

// ---------------------------------------------------------------------------
// Convert x (f32) -> Xh (f16), 4 elements per thread, b128 load / b64 store.
// ---------------------------------------------------------------------------
__global__ __launch_bounds__(256) void ffm_cvt(const float* __restrict__ x,
                                               _Float16* __restrict__ Xh) {
    size_t idx = ((size_t)blockIdx.x * 256 + threadIdx.x) * 4;
    v4f v = *(const v4f*)(x + idx);
    v4h o;
#pragma unroll
    for (int i = 0; i < 4; ++i) o[i] = (_Float16)v[i];
    *(v4h*)(Xh + idx) = o;
}

// ---------------------------------------------------------------------------
// Build B = A^T (B[j][i] = A[i][j], nonzero for i<j), emitted directly in the
// CDNA5 WMMA 16-bit B-fragment layout:
//   fragment (nt, kt): N = nt*16 + (lane&15), K = kt*32 + Kmap(lane, half)
//   Kmap: half h -> h + (h>=8 ? 8:0) + ((lane>=16) ? 8:0)
// Per-lane halves stored as two contiguous b128 groups (h0..7 then h8..15) so
// the consumer issues two fully-coalesced 16B loads per fragment. Fragments
// are ordered (nt*32 + kt), making each 4-nt x 32-kt panel 128KB contiguous —
// exactly one 1-D TDM descriptor.
// ---------------------------------------------------------------------------
__global__ __launch_bounds__(256) void ffm_buildB(const float* __restrict__ w,
                                                  const int* __restrict__ fields,
                                                  _Float16* __restrict__ Bp) {
    const int lane   = threadIdx.x & 31;
    const int wave   = threadIdx.x >> 5;
    const int fragId = blockIdx.x * 8 + wave;   // 0..2047 = nt*32 + kt
    const int nt     = fragId >> 5;             // 0..63
    const int kt     = fragId & 31;             // 0..31
    const int l16    = lane & 15;
    const int kAdd   = (lane >> 4) * 8;

    const int i  = nt * 16 + l16;               // N index (original i)
    const int fi = fields[i];
    const float* wi = w + (size_t)i * 32 * 8;   // w[i, :, :]

    v8h lo, hi;
#pragma unroll
    for (int h = 0; h < 16; ++h) {
        const int K = h + ((h >= 8) ? 8 : 0) + kAdd;
        const int j = kt * 32 + K;              // K index (original j)
        float val = 0.0f;
        if (i < j) {
            const int fj = fields[j];
            const float* a = wi + fj * 8;                   // w[i, fields[j], :]
            const float* b = w + ((size_t)j * 32 + fi) * 8; // w[j, fields[i], :]
            float s = 0.0f;
#pragma unroll
            for (int k = 0; k < 8; ++k) s += a[k] * b[k];
            val = s * 0.125f;                   // mean over K=8
        }
        if (h < 8) lo[h] = (_Float16)val;
        else       hi[h - 8] = (_Float16)val;
    }
    _Float16* frag = Bp + (size_t)fragId * 512;
    *(v8h*)(frag + lane * 8)       = lo;
    *(v8h*)(frag + 256 + lane * 8) = hi;
}

// ---------------------------------------------------------------------------
// GEMM + fused quadratic-form reduction:
//   T[b, n] = sum_k Xh[b, k] * B[k, n]      (v_wmma_f32_16x16x32_f16)
//   temp[b] += sum_n x[b, n] * T[b, n]      (epilogue, never materialize T)
// 8 waves per block, wave owns 16 rows. blockIdx.y selects a 16-N-tile slice.
// The slice's B panels (4 chunks x 128KB) are DMA'd into a 2x128KB LDS double
// buffer by the TDM (wave 0), prefetching chunk c+1 while all 8 waves consume
// chunk c from LDS — B is read from L2 once per block instead of once per
// wave (8x L2 traffic reduction, ~2 GB -> ~256 MB).
// ---------------------------------------------------------------------------
__global__ __launch_bounds__(256) void ffm_gemm(const _Float16* __restrict__ Xh,
                                                const _Float16* __restrict__ Bp,
                                                const float* __restrict__ x,
                                                float* __restrict__ temp) {
    __shared__ _Float16 ldsB[2][4 * 32 * 512];   // 2 x 128KB (<= 320KB WGP LDS)

    const int lane = threadIdx.x & 31;
    const int wave = threadIdx.x >> 5;
    const int m0   = blockIdx.x * 128 + wave * 16;
    const int l16  = lane & 15;
    const int hi16 = lane >> 4;
    const _Float16* xrow = Xh + (size_t)(m0 + l16) * DDIM;

    float rowAcc[8];
#pragma unroll
    for (int r = 0; r < 8; ++r) rowAcc[r] = 0.0f;

    const int ntBase = blockIdx.y * 16;          // 16 N-tiles per slice

    // Prefetch chunk 0 (TDM is a per-wave DMA; EXEC-independent, wave 0 only).
    if (wave == 0)
        tdm_load_1d((unsigned int)(size_t)&ldsB[0][0],
                    Bp + (size_t)(ntBase + 0) * 32 * 512, 16384u);

    for (int c = 0; c < 4; ++c) {
        if (wave == 0) __builtin_amdgcn_s_wait_tensorcnt(0);
        __syncthreads();                          // buffer c&1 ready for all
        if (c + 1 < 4 && wave == 0)
            tdm_load_1d((unsigned int)(size_t)&ldsB[(c + 1) & 1][0],
                        Bp + (size_t)(ntBase + (c + 1) * 4) * 32 * 512, 16384u);

        const _Float16* bbuf = &ldsB[c & 1][0];
        v8f acc[4] = {};
        for (int kt = 0; kt < 32; ++kt) {
            ABu a;
            const _Float16* ap = xrow + kt * 32 + hi16 * 8;
            a.p[0] = *(const v8h*)(ap);           // K = kAdd + 0..7
            a.p[1] = *(const v8h*)(ap + 16);      // K = kAdd + 16..23
#pragma unroll
            for (int u = 0; u < 4; ++u) {
                const _Float16* fp = bbuf + ((size_t)u * 32 + kt) * 512;
                ABu b;
                b.p[0] = *(const v8h*)(fp + lane * 8);        // ds_load_b128
                b.p[1] = *(const v8h*)(fp + 256 + lane * 8);  // ds_load_b128
                acc[u] = __builtin_amdgcn_wmma_f32_16x16x32_f16(
                    /*neg_a=*/false, a.v, /*neg_b=*/false, b.v,
                    /*c_mod=*/(short)0, acc[u],
                    /*reuse_a=*/false, /*reuse_b=*/false);
            }
        }
        // Epilogue: temp contribution. C/D layout: VGPR r, lane L ->
        // row M = r + (L>=16 ? 8:0), col N = L&15.
#pragma unroll
        for (int u = 0; u < 4; ++u) {
            const int n = (ntBase + c * 4 + u) * 16 + l16;
#pragma unroll
            for (int r = 0; r < 8; ++r) {
                const int row = m0 + hi16 * 8 + r;
                rowAcc[r] += acc[u][r] * x[(size_t)row * DDIM + n];
            }
        }
        __syncthreads();   // everyone done reading buffer before it's refilled
    }

    // Reduce across the 16 lanes of each half-wave (wave32), then accumulate.
#pragma unroll
    for (int r = 0; r < 8; ++r) {
        float v = rowAcc[r];
        v += __shfl_xor(v, 1, 32);
        v += __shfl_xor(v, 2, 32);
        v += __shfl_xor(v, 4, 32);
        v += __shfl_xor(v, 8, 32);
        if (l16 == 0) atomicAdd(&temp[m0 + hi16 * 8 + r], v);
    }
}

// ---------------------------------------------------------------------------
// linear = x @ fc_w + fc_b; out = sigmoid(linear + temp). One wave per row.
// ---------------------------------------------------------------------------
__global__ __launch_bounds__(256) void ffm_out(const float* __restrict__ x,
                                               const float* __restrict__ fc_w,
                                               const float* __restrict__ fc_b,
                                               const float* __restrict__ temp,
                                               float* __restrict__ out) {
    const int lane = threadIdx.x & 31;
    const int wave = threadIdx.x >> 5;
    const int row  = blockIdx.x * 8 + wave;
    const float* xr = x + (size_t)row * DDIM;
    float s = 0.0f;
#pragma unroll 8
    for (int t = 0; t < DDIM / 32; ++t)
        s += xr[lane + t * 32] * fc_w[lane + t * 32];
    s += __shfl_xor(s, 1, 32);
    s += __shfl_xor(s, 2, 32);
    s += __shfl_xor(s, 4, 32);
    s += __shfl_xor(s, 8, 32);
    s += __shfl_xor(s, 16, 32);
    if (lane == 0) {
        const float z = s + fc_b[0] + temp[row];
        out[row] = 1.0f / (1.0f + __expf(-z));
    }
}

// ---------------------------------------------------------------------------
extern "C" void kernel_launch(void* const* d_in, const int* in_sizes, int n_in,
                              void* d_out, int out_size, void* d_ws, size_t ws_size,
                              hipStream_t stream) {
    (void)in_sizes; (void)n_in; (void)out_size; (void)ws_size;
    const float* x      = (const float*)d_in[0];   // (B, D) f32
    const float* w      = (const float*)d_in[1];   // (D, F, K) f32
    const float* fc_w   = (const float*)d_in[2];   // (D, 1) f32
    const float* fc_b   = (const float*)d_in[3];   // (1,) f32
    const int*   fields = (const int*)d_in[4];     // (D,) int32
    float* out = (float*)d_out;                    // (B, 1) f32

    char* ws = (char*)d_ws;
    _Float16* Xh   = (_Float16*)ws;                                   // 32 MB
    _Float16* Bp   = (_Float16*)(ws + (size_t)32 * 1024 * 1024);      //  2 MB
    float*    temp = (float*)   (ws + (size_t)34 * 1024 * 1024);      // 64 KB

    ffm_zero<<<BDIM / 256, 256, 0, stream>>>(temp);
    ffm_cvt<<<(BDIM * DDIM) / (256 * 4), 256, 0, stream>>>(x, Xh);
    ffm_buildB<<<256, 256, 0, stream>>>(w, fields, Bp);
    ffm_gemm<<<dim3(BDIM / 128, 4), 256, 0, stream>>>(Xh, Bp, x, temp);
    ffm_out<<<BDIM / 8, 256, 0, stream>>>(x, fc_w, fc_b, temp, out);
}
